// PointnetSAModuleMSG_72052371357926
// MI455X (gfx1250) — matrix operand order repaired and checked by reading, not compiled, not run on hardware
//
#include <hip/hip_runtime.h>

#define BATCH   8
#define NPTS    8192
#define C_IN    64
#define NPOINT  1024

typedef __attribute__((ext_vector_type(16))) __bf16 v16bf;
typedef __attribute__((ext_vector_type(8)))  __bf16 v8bf;
typedef __attribute__((ext_vector_type(8)))  float  v8f;

// ---------------------------------------------------------------------------
// Weight prep: pad K to multiple of 32, convert to bf16, fold BN into (s, t).
// ---------------------------------------------------------------------------
__global__ void prep_layer(const float* __restrict__ w,
                           const float* __restrict__ gamma,
                           const float* __restrict__ beta,
                           const float* __restrict__ mean,
                           const float* __restrict__ var,
                           __bf16* __restrict__ wOut,
                           float* __restrict__ sOut, float* __restrict__ tOut,
                           int cout, int cin, int kpad) {
  int i = blockIdx.x * blockDim.x + threadIdx.x;
  int total = cout * kpad;
  if (i < total) {
    int o = i / kpad, k = i % kpad;
    float v = (k < cin) ? w[o * cin + k] : 0.0f;
    wOut[i] = (__bf16)v;
  }
  if (i < cout) {
    float s = gamma[i] * rsqrtf(var[i] + 1e-5f);
    sOut[i] = s;
    tOut[i] = beta[i] - mean[i] * s;
  }
}

// ---------------------------------------------------------------------------
// Furthest point sampling: one workgroup per batch, 1024 threads, 8 pts/thread
// kept in registers. Per-step argmax via wave shuffle + 32-entry LDS reduce.
// ---------------------------------------------------------------------------
__global__ __launch_bounds__(1024) void fps_kernel(const float* __restrict__ xyz,
                                                   int* __restrict__ fps_idx,
                                                   float* __restrict__ new_xyz) {
  const int b = blockIdx.x;
  const int t = threadIdx.x;
  const float* xb = xyz + (size_t)b * NPTS * 3;

  float px[8], py[8], pz[8], dist[8];
#pragma unroll
  for (int k = 0; k < 8; ++k) {
    int p = t + k * 1024;
    px[k] = xb[p * 3 + 0];
    py[k] = xb[p * 3 + 1];
    pz[k] = xb[p * 3 + 2];
    dist[k] = 1e10f;
  }

  __shared__ float bxs, bys, bzs;
  __shared__ float rv[32];
  __shared__ int   ri[32];
  const int lane = t & 31, wv = t >> 5;

  if (t == 0) {
    bxs = px[0]; bys = py[0]; bzs = pz[0];
    fps_idx[b * NPOINT] = 0;
    new_xyz[(size_t)(b * NPOINT) * 3 + 0] = px[0];
    new_xyz[(size_t)(b * NPOINT) * 3 + 1] = py[0];
    new_xyz[(size_t)(b * NPOINT) * 3 + 2] = pz[0];
  }

  for (int it = 1; it < NPOINT; ++it) {
    __syncthreads();
    float cx = bxs, cy = bys, cz = bzs;
    float best = -1.0f; int bidx = 0x7fffffff;
#pragma unroll
    for (int k = 0; k < 8; ++k) {
      float dx = px[k] - cx, dy = py[k] - cy, dz = pz[k] - cz;
      float d2 = dx * dx + dy * dy + dz * dz;
      dist[k] = fminf(dist[k], d2);
      int p = t + k * 1024;
      if (dist[k] > best || (dist[k] == best && p < bidx)) { best = dist[k]; bidx = p; }
    }
    // wave32 reduce (argmax, smallest index tiebreak)
#pragma unroll
    for (int off = 16; off > 0; off >>= 1) {
      float ov = __shfl_down(best, off);
      int   oi = __shfl_down(bidx, off);
      if (ov > best || (ov == best && oi < bidx)) { best = ov; bidx = oi; }
    }
    if (lane == 0) { rv[wv] = best; ri[wv] = bidx; }
    __syncthreads();
    if (t < 32) {
      best = rv[t]; bidx = ri[t];
#pragma unroll
      for (int off = 16; off > 0; off >>= 1) {
        float ov = __shfl_down(best, off);
        int   oi = __shfl_down(bidx, off);
        if (ov > best || (ov == best && oi < bidx)) { best = ov; bidx = oi; }
      }
      if (t == 0) {
        fps_idx[b * NPOINT + it] = bidx;
        const float* pp = xb + (size_t)bidx * 3;
        float nx = pp[0], ny = pp[1], nz = pp[2];
        bxs = nx; bys = ny; bzs = nz;
        size_t o = (size_t)(b * NPOINT + it) * 3;
        new_xyz[o + 0] = nx; new_xyz[o + 1] = ny; new_xyz[o + 2] = nz;
      }
    }
  }
}

// ---------------------------------------------------------------------------
// Ball query: one wave per center; sequential 32-wide scan with ballot
// compaction keeps ascending index order (== reference top_k semantics).
// ---------------------------------------------------------------------------
template <int NS>
__global__ __launch_bounds__(256) void ballq_kernel(const float* __restrict__ xyz,
                                                    const float* __restrict__ new_xyz,
                                                    int* __restrict__ out_idx,
                                                    float r2) {
  const int gw = (blockIdx.x * 256 + threadIdx.x) >> 5;
  const int lane = threadIdx.x & 31;
  if (gw >= BATCH * NPOINT) return;
  const int b = gw / NPOINT;
  const float* xb = xyz + (size_t)b * NPTS * 3;
  const float cx = new_xyz[(size_t)gw * 3 + 0];
  const float cy = new_xyz[(size_t)gw * 3 + 1];
  const float cz = new_xyz[(size_t)gw * 3 + 2];
  int* oi = out_idx + (size_t)gw * NS;

  int count = 0;
  int first = NPTS - 1;
  for (int base = 0; base < NPTS && count < NS; base += 32) {
    int p = base + lane;
    float dx = xb[p * 3 + 0] - cx;
    float dy = xb[p * 3 + 1] - cy;
    float dz = xb[p * 3 + 2] - cz;
    bool hit = (dx * dx + dy * dy + dz * dz) < r2;
    unsigned mask = (unsigned)__ballot(hit);
    if (count == 0 && mask) first = base + __ffs(mask) - 1;
    int pos = count + __popc(mask & ((1u << lane) - 1u));
    if (hit && pos < NS) oi[pos] = p;
    count += __popc(mask);
  }
  if (count < NS) {
    int pad = (count > 0) ? first : (NPTS - 1);
    for (int j = count + lane; j < NS; j += 32) oi[j] = pad;
  }
}

// ---------------------------------------------------------------------------
// One bf16 WMMA GEMM layer: out[M x Cout] = relu(in[M x K] * W^T * s + t).
// A fragments from LDS, 16 contiguous K per lane half (k = 16h + s); B fragments
// are 32B contiguous loads of the bf16 weight rows (column n = out channel).
// ---------------------------------------------------------------------------
__device__ __forceinline__ void gemm_layer(const __bf16* __restrict__ inBuf, int inStride,
                                           __bf16* __restrict__ outBuf, int outStride,
                                           const __bf16* __restrict__ W, int kdim,
                                           const float* __restrict__ S,
                                           const float* __restrict__ T,
                                           int M, int Cout, int tid) {
  const int lane = tid & 31;
  const int wave = tid >> 5;
  const int col = lane & 15;     // row of A / column of tile
  const int hi = lane >> 4;      // lane group selects K sub-chunks
  const int Mt = M >> 4, Nt = Cout >> 4;

  for (int tile = wave; tile < Mt * Nt; tile += 8) {
    const int mi = tile / Nt, ni = tile % Nt;
    v8f acc = {};
    const __bf16* arow = inBuf + (size_t)(mi * 16 + col) * inStride;
    const __bf16* wrow = W + (size_t)(ni * 16 + col) * kdim;
    for (int kk = 0; kk < kdim; kk += 32) {
      // A: lanes 0-15 hold K kk..kk+15; lanes 16-31 hold K kk+16..kk+31.
      v16bf a = *(const v16bf*)(arow + kk + hi * 16);
      // B: lane holds 16 contiguous-K values of its output channel.
      v16bf bf = *(const v16bf*)(wrow + kk + hi * 16);
      acc = __builtin_amdgcn_wmma_f32_16x16x32_bf16(
          false, a, false, bf, (short)0, acc, false, false);
    }
    const int n = ni * 16 + col;
    const float sc = S[n], tt = T[n];
#pragma unroll
    for (int v = 0; v < 8; ++v) {
      float r = acc[v] * sc + tt;          // folded inference BN
      r = r > 0.0f ? r : 0.0f;             // ReLU
      outBuf[(size_t)(mi * 16 + v + hi * 8) * outStride + n] = (__bf16)r;
    }
  }
}

// ---------------------------------------------------------------------------
// Per-center grouped MLP + maxpool. One workgroup (8 waves) per center.
// ---------------------------------------------------------------------------
template <int NS, int C1, int C2, int C3, int SB>
__global__ __launch_bounds__(256) void mlp_kernel(
    const float* __restrict__ xyz, const float* __restrict__ points,
    const float* __restrict__ new_xyz, const int* __restrict__ ballidx,
    const __bf16* __restrict__ W1, const __bf16* __restrict__ W2,
    const __bf16* __restrict__ W3,
    const float* __restrict__ S1, const float* __restrict__ T1,
    const float* __restrict__ S2, const float* __restrict__ T2,
    const float* __restrict__ S3, const float* __restrict__ T3,
    float* __restrict__ outFeat, int outOff) {
  __shared__ __align__(32) __bf16 bufIn[NS * 96];
  __shared__ __align__(32) __bf16 bufOut[NS * SB];

  const int g = blockIdx.x;  // b*NPOINT + s
  const int b = g / NPOINT;
  const int tid = threadIdx.x;
  const int* idx = ballidx + (size_t)g * NS;
  const float cx = new_xyz[(size_t)g * 3 + 0];
  const float cy = new_xyz[(size_t)g * 3 + 1];
  const float cz = new_xyz[(size_t)g * 3 + 2];
  const float* xb = xyz + (size_t)b * NPTS * 3;
  const float* pb = points + (size_t)b * NPTS * C_IN;

  __builtin_prefetch(W1 + (size_t)tid * 16, 0, 1);  // warm first weight tiles

  // Gather [rel_xyz | point feats | zero pad] -> bf16 LDS tile (NS x 96).
  for (int e = tid; e < NS * 96; e += 256) {
    int m = e / 96, k = e % 96;
    int p = idx[m];
    float v;
    if (k < 3) {
      v = xb[(size_t)p * 3 + k] - (k == 0 ? cx : (k == 1 ? cy : cz));
    } else if (k < 67) {
      v = pb[(size_t)p * C_IN + (k - 3)];
    } else {
      v = 0.0f;
    }
    bufIn[e] = (__bf16)v;
  }
  __syncthreads();
  gemm_layer(bufIn, 96, bufOut, SB, W1, 96, S1, T1, NS, C1, tid);
  __syncthreads();
  gemm_layer(bufOut, SB, bufIn, 96, W2, C1, S2, T2, NS, C2, tid);
  __syncthreads();
  gemm_layer(bufIn, 96, bufOut, SB, W3, C2, S3, T3, NS, C3, tid);
  __syncthreads();

  // Maxpool over the NS neighbors (everything is >= 0 post-ReLU).
  for (int n = tid; n < C3; n += 256) {
    float mx = -1e30f;
    for (int m = 0; m < NS; ++m) {
      float v = (float)bufOut[(size_t)m * SB + n];
      mx = v > mx ? v : mx;
    }
    outFeat[(size_t)g * 320 + outOff + n] = mx;
  }
}

// ---------------------------------------------------------------------------
// Host launcher.
// d_in: [xyz, points, then per (scale,layer) sorted-key leaves:
//        beta, gamma, mean, var, w]  (jax dict flattening order)
// d_out: new_xyz (B*1024*3 f32) ++ features (B*1024*320 f32)
// ---------------------------------------------------------------------------
extern "C" void kernel_launch(void* const* d_in, const int* in_sizes, int n_in,
                              void* d_out, int out_size, void* d_ws, size_t ws_size,
                              hipStream_t stream) {
  (void)in_sizes; (void)n_in; (void)out_size; (void)ws_size;
  const float* xyz = (const float*)d_in[0];
  const float* points = (const float*)d_in[1];
  auto P = [&](int scale, int layer, int which) -> const float* {
    return (const float*)d_in[2 + (scale * 3 + layer) * 5 + which];
  };  // which: 0=beta 1=gamma 2=mean 3=var 4=w

  static const int cins[9]  = {67, 32, 32, 67, 64, 64, 67, 64, 96};
  static const int couts[9] = {32, 32, 64, 64, 64, 128, 64, 96, 128};
  static const int kpads[9] = {96, 32, 32, 96, 64, 64, 96, 64, 96};

  char* ws = (char*)d_ws;
  int* fps_idx = (int*)ws;  ws += (size_t)BATCH * NPOINT * 4;
  int* bq0 = (int*)ws;      ws += (size_t)BATCH * NPOINT * 16 * 4;
  int* bq1 = (int*)ws;      ws += (size_t)BATCH * NPOINT * 32 * 4;
  int* bq2 = (int*)ws;      ws += (size_t)BATCH * NPOINT * 128 * 4;
  ws = (char*)(((uintptr_t)ws + 31) & ~(uintptr_t)31);
  void* wbf[9]; float* sArr[9]; float* tArr[9];
  for (int l = 0; l < 9; ++l) { wbf[l] = ws; ws += (size_t)couts[l] * kpads[l] * 2; }
  ws = (char*)(((uintptr_t)ws + 15) & ~(uintptr_t)15);
  for (int l = 0; l < 9; ++l) {
    sArr[l] = (float*)ws; ws += (size_t)couts[l] * 4;
    tArr[l] = (float*)ws; ws += (size_t)couts[l] * 4;
  }

  float* new_xyz = (float*)d_out;
  float* feat = (float*)d_out + (size_t)BATCH * NPOINT * 3;

  for (int l = 0; l < 9; ++l) {
    int sc = l / 3, ly = l % 3;
    int total = couts[l] * kpads[l];
    prep_layer<<<(total + 255) / 256, 256, 0, stream>>>(
        P(sc, ly, 4), P(sc, ly, 1), P(sc, ly, 0), P(sc, ly, 2), P(sc, ly, 3),
        (__bf16*)wbf[l], sArr[l], tArr[l], couts[l], cins[l], kpads[l]);
  }

  fps_kernel<<<BATCH, 1024, 0, stream>>>(xyz, fps_idx, new_xyz);

  const int bqBlocks = (BATCH * NPOINT) / 8;  // 8 waves per 256-thread block
  ballq_kernel<16><<<bqBlocks, 256, 0, stream>>>(xyz, new_xyz, bq0, 0.1f * 0.1f);
  ballq_kernel<32><<<bqBlocks, 256, 0, stream>>>(xyz, new_xyz, bq1, 0.2f * 0.2f);
  ballq_kernel<128><<<bqBlocks, 256, 0, stream>>>(xyz, new_xyz, bq2, 0.4f * 0.4f);

  const int G = BATCH * NPOINT;
  mlp_kernel<16, 32, 32, 64, 64><<<G, 256, 0, stream>>>(
      xyz, points, new_xyz, bq0,
      (const __bf16*)wbf[0], (const __bf16*)wbf[1], (const __bf16*)wbf[2],
      sArr[0], tArr[0], sArr[1], tArr[1], sArr[2], tArr[2], feat, 0);
  mlp_kernel<32, 64, 64, 128, 128><<<G, 256, 0, stream>>>(
      xyz, points, new_xyz, bq1,
      (const __bf16*)wbf[3], (const __bf16*)wbf[4], (const __bf16*)wbf[5],
      sArr[3], tArr[3], sArr[4], tArr[4], sArr[5], tArr[5], feat, 64);
  mlp_kernel<128, 64, 96, 128, 128><<<G, 256, 0, stream>>>(
      xyz, points, new_xyz, bq2,
      (const __bf16*)wbf[6], (const __bf16*)wbf[7], (const __bf16*)wbf[8],
      sArr[6], tArr[6], sArr[7], tArr[7], sArr[8], tArr[8], feat, 192);
}
